// PGD_84731114815682
// MI455X (gfx1250) — compile-verified, matrix operand order checked by reading, not run on hardware
//
#include <hip/hip_runtime.h>
#include <hip/hip_bf16.h>

// ---------------------------------------------------------------------------
// Problem constants (from reference): B=2, S=1024, T=S+1, V=32000, D=768, H=12
// All K dims used are multiples of 32; all N dims are multiples of 128.
// Only M can be ragged (T=1025, final logits M=2).
// ---------------------------------------------------------------------------
static constexpr int Bc = 2;
static constexpr int S  = 1024;
static constexpr int T  = 1025;
static constexpr int V  = 32000;
static constexpr int D  = 768;
static constexpr int H  = 12;
static constexpr int NL = 6;
static constexpr float INV_N = 1.0f / 1024.0f;

typedef __attribute__((ext_vector_type(16))) __bf16 v16bf;
typedef __attribute__((ext_vector_type(8)))  float  v8f;

static constexpr int BM = 128;     // block tile M
static constexpr int BN = 128;     // block tile N (all N are multiples of 128)
static constexpr int BK = 32;      // one WMMA K-step (all K are multiples of 32)
static constexpr int LDP = BK + 8; // padded LDS row (40 ushorts = 80B, 16B-aligned)

__device__ __forceinline__ unsigned short f2bf(float f) {
  union { float f; unsigned u; } x; x.f = f;
  unsigned r = x.u + 0x7FFFu + ((x.u >> 16) & 1u);   // RNE
  return (unsigned short)(r >> 16);
}

union V16U { v16bf v; unsigned short u[16]; uint4 q[2]; };
union U4x8 { uint4 q; unsigned short u[8]; };

// ---------------------------------------------------------------------------
// bf16 WMMA GEMM with double-buffered LDS + register-relayed global prefetch.
//   C(MxN) = A(MxK) * op(B)
//   TRANSB=1: B stored as N x K (row n contiguous in k)  -> fully vectorized
//   TRANSB=0: B stored as K x N                          -> vector load + LDS scatter
// EPI 0: Cf = acc       EPI 1: Cf += acc
// EPI 2: Cbf = bf16(exp(acc)), denom[m] += exp(acc)
// EPI 3: Cbf = bf16(acc * scale[idx])  (scale=1 if ptr null)
// ---------------------------------------------------------------------------
template <int EPI, bool TRANSB>
__global__ __launch_bounds__(256) void gemm_bf16(
    const unsigned short* __restrict__ A,
    const unsigned short* __restrict__ Bm,
    float* __restrict__ Cf,
    unsigned short* __restrict__ Cbf,
    float* __restrict__ denom,
    const float* __restrict__ scale_ptr, int scale_idx,
    int M, int N, int Kd, int lda, int ldb, int ldc)
{
  __shared__ __align__(16) unsigned short As[2][BM][LDP];  // A tiles, row-major
  __shared__ __align__(16) unsigned short Bs[2][BN][LDP];  // B tiles, TRANSPOSED: [n][k]
  __shared__ float rsum[BM];

  const int tid  = threadIdx.x;
  const int lane = tid & 31;
  const int wave = tid >> 5;
  const int wm   = wave & 3;     // 4 waves along M (4*32 = 128)
  const int wn   = wave >> 2;    // 2 waves along N (2*64 = 128)
  const int m0   = blockIdx.y * BM;
  const int n0   = blockIdx.x * BN;

  const int half = lane >> 4;    // 0: lanes 0-15, 1: lanes 16-31
  const int l15  = lane & 15;
  const int kb   = half * 8;     // K sub-base within 16-wide half (frag layout)

  // ---- per-thread staging coordinates (fixed across K-steps) ----
  const int t2[2] = { tid, tid + 256 };

  // load tile (k0) into registers: 2x uint4 for A, 2x uint4 for B
  auto load_tile = [&](int k0, uint4* ra, uint4* rb) {
#pragma unroll
    for (int i = 0; i < 2; ++i) {
      int r = t2[i] >> 2, cg = t2[i] & 3;
      int gm = m0 + r;
      ra[i] = (gm < M) ? *(const uint4*)&A[(long)gm * lda + k0 + cg * 8]
                       : make_uint4(0u, 0u, 0u, 0u);
    }
    if (TRANSB) {
#pragma unroll
      for (int i = 0; i < 2; ++i) {
        int n = t2[i] >> 2, kg = t2[i] & 3;
        rb[i] = *(const uint4*)&Bm[(long)(n0 + n) * ldb + k0 + kg * 8];
      }
    } else {
#pragma unroll
      for (int i = 0; i < 2; ++i) {
        int k = t2[i] >> 4, ng = t2[i] & 15;
        rb[i] = *(const uint4*)&Bm[(long)(k0 + k) * ldb + n0 + ng * 8];
      }
    }
  };
  // drain registers into LDS buffer `buf`
  auto store_tile = [&](int buf, const uint4* ra, const uint4* rb) {
#pragma unroll
    for (int i = 0; i < 2; ++i) {
      int r = t2[i] >> 2, cg = t2[i] & 3;
      *(uint4*)&As[buf][r][cg * 8] = ra[i];
    }
    if (TRANSB) {
#pragma unroll
      for (int i = 0; i < 2; ++i) {
        int n = t2[i] >> 2, kg = t2[i] & 3;
        *(uint4*)&Bs[buf][n][kg * 8] = rb[i];
      }
    } else {
#pragma unroll
      for (int i = 0; i < 2; ++i) {
        int k = t2[i] >> 4, ng = t2[i] & 15;
        U4x8 val; val.q = rb[i];
#pragma unroll
        for (int j = 0; j < 8; ++j) Bs[buf][ng * 8 + j][k] = val.u[j];
      }
    }
  };

  v8f zero;
#pragma unroll
  for (int t = 0; t < 8; ++t) zero[t] = 0.0f;
  v8f acc[2][4];
#pragma unroll
  for (int i = 0; i < 2; ++i)
#pragma unroll
    for (int j = 0; j < 4; ++j) acc[i][j] = zero;

  uint4 ra[2], rb[2];
  load_tile(0, ra, rb);
  store_tile(0, ra, rb);
  __syncthreads();

  const int nsteps = Kd / BK;    // all K dims are multiples of BK
  for (int s = 0; s < nsteps; ++s) {
    const int cur  = s & 1;
    const bool more = (s + 1 < nsteps);
    // issue global loads for tile s+1 (latency hidden under compute below)
    if (more) load_tile((s + 1) * BK, ra, rb);
    // L2 prefetch for tile s+2
    if (s + 2 < nsteps) {
      int pr = m0 + (tid & (BM - 1));
      if (pr < M) __builtin_prefetch(&A[(long)pr * lda + (s + 2) * BK]);
    }

    // ---- compute tile s from LDS: K-contiguous halves -> 2x ds_load_b128 ----
    V16U af[2], bfm[4];
#pragma unroll
    for (int i = 0; i < 2; ++i) {
      int row = wm * 32 + i * 16 + l15;
      af[i].q[0] = *(const uint4*)&As[cur][row][kb];
      af[i].q[1] = *(const uint4*)&As[cur][row][16 + kb];
    }
#pragma unroll
    for (int j = 0; j < 4; ++j) {
      int col = wn * 64 + j * 16 + l15;
      bfm[j].q[0] = *(const uint4*)&Bs[cur][col][kb];
      bfm[j].q[1] = *(const uint4*)&Bs[cur][col][16 + kb];
    }
#pragma unroll
    for (int i = 0; i < 2; ++i)
#pragma unroll
      for (int j = 0; j < 4; ++j)
        acc[i][j] = __builtin_amdgcn_wmma_f32_16x16x32_bf16(
            false, af[i].v, false, bfm[j].v, (short)0, acc[i][j], false, false);

    // drain prefetched registers into the other buffer, then one barrier
    if (more) store_tile(cur ^ 1, ra, rb);
    __syncthreads();
  }

  // ---- epilogue ----
  if (EPI == 2) {
    for (int r = tid; r < BM; r += 256) rsum[r] = 0.0f;
    __syncthreads();
  }
  float scale = 1.0f;
  if (EPI == 3 && scale_ptr) scale = scale_ptr[scale_idx];

#pragma unroll
  for (int i = 0; i < 2; ++i)
#pragma unroll
    for (int j = 0; j < 4; ++j) {
      int n = n0 + wn * 64 + j * 16 + l15;
#pragma unroll
      for (int r = 0; r < 8; ++r) {
        int ml = wm * 32 + i * 16 + r + half * 8;  // C/D layout: VGPR r -> M=r / M=8+r
        int m = m0 + ml;
        if (m < M) {
          float v = acc[i][j][r];
          long  o = (long)m * ldc + n;
          if (EPI == 0) {
            Cf[o] = v;
          } else if (EPI == 1) {
            Cf[o] += v;
          } else if (EPI == 2) {
            float ev = __expf(v);
            Cbf[o] = f2bf(ev);
            atomicAdd(&rsum[ml], ev);
          } else {
            Cbf[o] = f2bf(v * scale);
          }
        }
      }
    }

  if (EPI == 2) {
    __syncthreads();
    for (int r = tid; r < BM; r += 256) {
      int m = m0 + r;
      if (m < M) atomicAdd(&denom[m], rsum[r]);
    }
  }
}

template <int EPI, bool TB>
static void launch_gemm(hipStream_t st, const unsigned short* A, const unsigned short* Bm,
                        float* Cf, unsigned short* Cbf, float* den,
                        const float* sp, int si,
                        int M, int N, int K, int lda, int ldb, int ldc) {
  dim3 g((N + BN - 1) / BN, (M + BM - 1) / BM);
  gemm_bf16<EPI, TB><<<g, 256, 0, st>>>(A, Bm, Cf, Cbf, den, sp, si, M, N, K, lda, ldb, ldc);
}

// ---------------------------------------------------------------------------
// Elementwise helpers
// ---------------------------------------------------------------------------
__global__ void k_cvt(const float* __restrict__ s, unsigned short* __restrict__ d, long n) {
  long i = (long)blockIdx.x * 256 + threadIdx.x;
  long st = (long)gridDim.x * 256;
  for (; i < n; i += st) d[i] = f2bf(s[i]);
}
// W_e (V x D) -> bf16 transpose (D x V)
__global__ void k_transposeWe(const float* __restrict__ We, unsigned short* __restrict__ Wet,
                              long n) {
  long i = (long)blockIdx.x * 256 + threadIdx.x;
  long st = (long)gridDim.x * 256;
  for (; i < n; i += st) {
    long d = i / V, v = i % V;
    Wet[i] = f2bf(We[v * (long)D + d]);
  }
}
__global__ void k_zero(float* __restrict__ p, long n) {
  long i = (long)blockIdx.x * 256 + threadIdx.x;
  long st = (long)gridDim.x * 256;
  for (; i < n; i += st) p[i] = 0.0f;
}
__global__ void k_gather(const int* __restrict__ idx, const float* __restrict__ We,
                         float* __restrict__ e, long n) {
  long i = (long)blockIdx.x * 256 + threadIdx.x;
  long st = (long)gridDim.x * 256;
  for (; i < n; i += st) {
    long m = i / D, d = i % D;
    e[i] = We[(long)idx[m] * D + d];
  }
}
// f_k (B,T,D) rows s<S -> contiguous (B*S, D) bf16
__global__ void k_cvt_fk(const float* __restrict__ fk, unsigned short* __restrict__ o, long n) {
  long i = (long)blockIdx.x * 256 + threadIdx.x;
  long st = (long)gridDim.x * 256;
  const long SD = (long)S * D, TD = (long)T * D;
  for (; i < n; i += st) {
    long b = i / SD, rem = i % SD;
    o[i] = f2bf(fk[b * TD + rem]);
  }
}
__global__ void k_diff(const float* __restrict__ e, const float* __restrict__ E,
                       const float* __restrict__ denom, float* __restrict__ df,
                       unsigned short* __restrict__ dbf, long n) {
  long i = (long)blockIdx.x * 256 + threadIdx.x;
  long st = (long)gridDim.x * 256;
  for (; i < n; i += st) {
    long m = i / D;
    float den = denom[m] + 1e-8f;
    float v = e[i] - E[i] / den;
    df[i] = v;
    dbf[i] = f2bf(v);
  }
}
__global__ void k_dB(const float* __restrict__ df, const float* __restrict__ blr,
                     float* __restrict__ dB) {
  int i = blockIdx.x * 256 + threadIdx.x;
  if (i >= Bc * D) return;
  int b = i / D, d = i - b * D;
  float s = 0.0f;
  for (int ss = 0; ss < S; ++ss) s += df[((long)b * S + ss) * D + d];
  dB[i] = s * blr[0];
}
__global__ void k_update(float* __restrict__ fk, const float* __restrict__ dA,
                         const float* __restrict__ dB, long n) {
  long i = (long)blockIdx.x * 256 + threadIdx.x;
  long st = (long)gridDim.x * 256;
  const long TD = (long)T * D;
  for (; i < n; i += st) {
    long b = i / TD;
    int d = (int)(i % D);
    fk[i] += INV_N * (dA[i] + dB[b * D + d]);
  }
}
__global__ void k_cvt_last(const float* __restrict__ fk, unsigned short* __restrict__ o) {
  int i = blockIdx.x * 256 + threadIdx.x;
  if (i >= Bc * D) return;
  int b = i / D, d = i - b * D;
  o[i] = f2bf(fk[((long)b * T + (T - 1)) * D + d]);
}

static inline int ewb(long n) { return (int)((n + 255) / 256); }

// ---------------------------------------------------------------------------
// kernel_launch: full pipeline on `stream`, graph-capture safe
// ---------------------------------------------------------------------------
extern "C" void kernel_launch(void* const* d_in, const int* in_sizes, int n_in,
                              void* d_out, int out_size, void* d_ws, size_t ws_size,
                              hipStream_t stream) {
  (void)in_sizes; (void)n_in; (void)out_size; (void)ws_size;
  const int*   idx  = (const int*)d_in[0];
  const float* W_e  = (const float*)d_in[1];
  const float* W_p  = (const float*)d_in[2];
  const float* W_k  = (const float*)d_in[3];
  const float* W_q  = (const float*)d_in[4];
  const float* W_v  = (const float*)d_in[5];
  const float* A_LR = (const float*)d_in[6];
  const float* B_LR = (const float*)d_in[7];
  float* out = (float*)d_out;

  char*  w   = (char*)d_ws;
  size_t off = 0;
  auto alloc = [&](size_t bytes) -> void* {
    off = (off + 255) & ~(size_t)255;
    void* p = w + off;
    off += bytes;
    return p;
  };
  const long DD = (long)D * D, SD = (long)S * D, TD = (long)T * D;

  unsigned short* We_bf   = (unsigned short*)alloc((long)V * D * 2);  // V x D
  unsigned short* Wet_bf  = (unsigned short*)alloc((long)V * D * 2);  // D x V (transposed)
  unsigned short* Wk_bf   = (unsigned short*)alloc((long)H * DD * 2);
  unsigned short* Wq_bf   = (unsigned short*)alloc((long)H * DD * 2);
  unsigned short* Wv_bf   = (unsigned short*)alloc((long)H * DD * 2);
  unsigned short* p_bf    = (unsigned short*)alloc(TD * 2);
  unsigned short* xi_bf   = (unsigned short*)alloc((long)H * SD * 2);
  unsigned short* xj_bf   = (unsigned short*)alloc((long)H * TD * 2);
  unsigned short* K_bf    = (unsigned short*)alloc((long)H * T * S * 2);
  float*          e_f     = (float*)alloc((long)Bc * SD * 4);
  float*          fk      = (float*)alloc((long)Bc * TD * 4);
  unsigned short* fk_bf   = (unsigned short*)alloc((long)Bc * SD * 2);
  unsigned short* Z_bf    = (unsigned short*)alloc((long)Bc * S * V * 2);
  float*          denomv  = (float*)alloc((long)Bc * S * 4);
  float*          E_f     = (float*)alloc((long)Bc * SD * 4);
  float*          diff_f  = (float*)alloc((long)Bc * SD * 4);
  unsigned short* diff_bf = (unsigned short*)alloc((long)Bc * SD * 2);
  float*          dBv     = (float*)alloc((long)Bc * D * 4);
  float*          dAv     = (float*)alloc((long)Bc * TD * 4);
  unsigned short* last_bf = (unsigned short*)alloc((long)Bc * D * 2);

  // ---- precompute: bf16 casts, W_e transpose, gather, f_k = 0 ----
  k_cvt<<<ewb((long)V * D), 256, 0, stream>>>(W_e, We_bf, (long)V * D);
  k_transposeWe<<<ewb((long)V * D), 256, 0, stream>>>(W_e, Wet_bf, (long)V * D);
  k_cvt<<<ewb((long)H * DD), 256, 0, stream>>>(W_k, Wk_bf, (long)H * DD);
  k_cvt<<<ewb((long)H * DD), 256, 0, stream>>>(W_q, Wq_bf, (long)H * DD);
  k_cvt<<<ewb((long)H * DD), 256, 0, stream>>>(W_v, Wv_bf, (long)H * DD);
  k_cvt<<<ewb(TD), 256, 0, stream>>>(W_p, p_bf, TD);
  k_gather<<<ewb((long)Bc * SD), 256, 0, stream>>>(idx, W_e, e_f, (long)Bc * SD);
  k_zero<<<ewb((long)Bc * TD), 256, 0, stream>>>(fk, (long)Bc * TD);

  // x_i[h] = p[:S] @ W_k[h];  x_j[h] = p[:T] @ W_q[h]
  for (int h = 0; h < H; ++h) {
    launch_gemm<3, false>(stream, p_bf, Wk_bf + h * DD, nullptr, xi_bf + h * SD,
                          nullptr, nullptr, 0, S, D, D, D, D, D);
    launch_gemm<3, false>(stream, p_bf, Wq_bf + h * DD, nullptr, xj_bf + h * TD,
                          nullptr, nullptr, 0, T, D, D, D, D, D);
  }
  // K[h] = x_j[h] @ x_i[h]^T  (TRANSB fast path)
  for (int h = 0; h < H; ++h) {
    launch_gemm<3, true>(stream, xj_bf + h * TD, xi_bf + h * SD, nullptr,
                         K_bf + (long)h * T * S, nullptr, nullptr, 0,
                         T, S, D, D, D, S);
  }

  // ---- 6 GD steps ----
  for (int layer = 0; layer < NL; ++layer) {
    k_cvt_fk<<<ewb((long)Bc * SD), 256, 0, stream>>>(fk, fk_bf, (long)Bc * SD);
    k_zero<<<ewb((long)Bc * S), 256, 0, stream>>>(denomv, (long)Bc * S);

    // Z = exp(f_k[:, :S] @ W_e^T), denom = row-sums   (TRANSB, B = W_e (VxD))
    launch_gemm<2, true>(stream, fk_bf, We_bf, nullptr, Z_bf, denomv,
                         nullptr, 0, Bc * S, V, D, D, D, V);
    // E = Z @ W_e   (TRANSB with transposed W_e (DxV))
    launch_gemm<0, true>(stream, Z_bf, Wet_bf, E_f, nullptr, nullptr,
                         nullptr, 0, Bc * S, D, V, V, V, D);
    // diff = e - E/denom
    k_diff<<<ewb((long)Bc * SD), 256, 0, stream>>>(e_f, E_f, denomv, diff_f,
                                                   diff_bf, (long)Bc * SD);
    // dB = B_LR * sum_s diff
    k_dB<<<ewb((long)Bc * D), 256, 0, stream>>>(diff_f, B_LR, dBv);
    // dA = 0
    k_zero<<<ewb((long)Bc * TD), 256, 0, stream>>>(dAv, (long)Bc * TD);

    for (int b = 0; b < Bc; ++b) {
      // Vh[b,h] = (diff[b] @ W_v[h]) * A_LR[h]  -> bf16 (stored in Z_bf region, Z dead now)
      for (int h = 0; h < H; ++h) {
        unsigned short* Vh_bh = Z_bf + ((long)(b * H + h)) * SD;
        launch_gemm<3, false>(stream, diff_bf + (long)b * SD, Wv_bf + h * DD,
                              nullptr, Vh_bh, nullptr, A_LR, h,
                              S, D, D, D, D, D);
      }
      // dA[b] += K[h] @ Vh[b,h]  over h
      for (int h = 0; h < H; ++h) {
        unsigned short* Vh_bh = Z_bf + ((long)(b * H + h)) * SD;
        launch_gemm<1, false>(stream, K_bf + (long)h * T * S, Vh_bh,
                              dAv + (long)b * TD, nullptr, nullptr, nullptr, 0,
                              T, D, S, S, D, D);
      }
    }
    // f_k += (dA + dB) / N
    k_update<<<ewb((long)Bc * TD), 256, 0, stream>>>(fk, dAv, dBv, (long)Bc * TD);
  }

  // ---- logits = f_k[:, -1] @ W_e^T ----
  k_cvt_last<<<ewb((long)Bc * D), 256, 0, stream>>>(fk, last_bf);
  launch_gemm<0, true>(stream, last_bf, We_bf, out, nullptr, nullptr,
                       nullptr, 0, Bc, V, D, D, D, V);
}
